// RotationAdapter_62096637165726
// MI455X (gfx1250) — matrix operand order repaired and checked by reading, not compiled, not run on hardware
//
#include <hip/hip_runtime.h>

// Reference: N=1048576 rows, D=128, 3-bit codebook (8 centroids, 7 boundaries).
// out = ((centroids[bucketize((k/||k|| @ R^T) * s)] * 1/s) @ R) * ||k||
//
// Scales are folded into the LDS rotation tables:
//   Rh [n][k] = R[n][k] * cs[n]      (GEMM1 directly produces "shaped")
//   RhT[j][n] = R[n][j] * inv[n]     (GEMM2 consumes raw centroids)

#define D      128
#define NCENT  8
#define WAVES  8                     // waves per block
#define ROWS_PER_TILE (WAVES * 16)   // 128 rows per tile
#define TILES_PER_BLOCK 8            // amortize rotation preload

typedef __attribute__((ext_vector_type(16))) _Float16 v16h;
typedef __attribute__((ext_vector_type(8)))  float    v8f;

union Frag16 {            // one WMMA 16-bit A/B fragment (8 VGPRs)
    v16h     v;
    uint4    u[2];
    _Float16 h[16];
};
union HalfPack8 {         // 8 halves == one b128 LDS store
    uint4    u;
    _Float16 h[8];
};

__global__ __launch_bounds__(256) void rotquant_fused_kernel(
    const float* __restrict__ k,
    const float* __restrict__ rot,        // R, row-major DxD
    const float* __restrict__ cscale,     // D
    const float* __restrict__ cent,       // 8
    float* __restrict__ out)
{
    __shared__ _Float16 Rh [D * D];            // cs-scaled R, row-major (32 KB)
    __shared__ _Float16 RhT[D * D];            // inv-scaled R^T         (32 KB)
    __shared__ _Float16 Qs[WAVES][16 * D];     // centroid tiles         (32 KB)
    __shared__ float    Nrm[WAVES][16];        // per-row norms

    const int tid  = threadIdx.x;
    const int wid  = tid >> 5;
    const int lane = tid & 31;
    const int m16  = lane & 15;     // A/B/C fragment "low" coordinate
    const int hi   = lane >> 4;     // lane half selector

    const size_t blk0 = (size_t)blockIdx.x * TILES_PER_BLOCK * ROWS_PER_TILE;

    // Kick HBM early for this wave's first rows while we build the LDS tables.
    __builtin_prefetch(k + (blk0 + wid * 16 + m16) * D, 0, 0);
    __builtin_prefetch(k + (blk0 + wid * 16 + m16) * D + 64, 0, 0);

    // ------- rotation -> LDS (f16, two orientations, scales folded in) -------
    {
        const int base = tid * 64;                     // 256 thr * 64 = D*D
        const int r    = tid >> 1;                     // row of R this chunk is in
        const int c0   = (tid & 1) * 64;
        const float csr  = cscale[r];                  // scale for output col r
        const float invr = 1.0f / fmaxf(csr, 1e-6f);   // inverse for contraction r
        const float4* rp = (const float4*)(rot + base);
#pragma unroll
        for (int j = 0; j < 8; ++j) {
            float4 a = rp[2 * j];
            float4 b = rp[2 * j + 1];
            HalfPack8 p;                               // Rh row r, scaled by cs[r]
            p.h[0] = (_Float16)(a.x * csr); p.h[1] = (_Float16)(a.y * csr);
            p.h[2] = (_Float16)(a.z * csr); p.h[3] = (_Float16)(a.w * csr);
            p.h[4] = (_Float16)(b.x * csr); p.h[5] = (_Float16)(b.y * csr);
            p.h[6] = (_Float16)(b.z * csr); p.h[7] = (_Float16)(b.w * csr);
            ((uint4*)(Rh + base))[j] = p.u;            // row-major, b128 store
            HalfPack8 q;                               // R^T col r, scaled by inv[r]
            q.h[0] = (_Float16)(a.x * invr); q.h[1] = (_Float16)(a.y * invr);
            q.h[2] = (_Float16)(a.z * invr); q.h[3] = (_Float16)(a.w * invr);
            q.h[4] = (_Float16)(b.x * invr); q.h[5] = (_Float16)(b.y * invr);
            q.h[6] = (_Float16)(b.z * invr); q.h[7] = (_Float16)(b.w * invr);
            const int c = c0 + j * 8;                  // transposed scatter
#pragma unroll
            for (int t = 0; t < 8; ++t)
                RhT[(c + t) * D + r] = q.h[t];
        }
    }

    // ------- loop-invariant per-lane constants: centroids, boundaries -------
    float cents[NCENT], bnd[NCENT - 1];
#pragma unroll
    for (int i = 0; i < NCENT; ++i) cents[i] = cent[i];
#pragma unroll
    for (int i = 0; i < NCENT - 1; ++i) bnd[i] = 0.5f * (cents[i] + cents[i + 1]);

    __syncthreads();                             // rotation tables ready

    // =================== loop over this block's consecutive row tiles ===================
#pragma unroll 1
    for (int t = 0; t < TILES_PER_BLOCK; ++t) {
        const size_t row0 = blk0 + (size_t)t * ROWS_PER_TILE + wid * 16;
        const float* krow = k + (row0 + m16) * D;

        if (t + 1 < TILES_PER_BLOCK) {           // stay ahead of HBM latency
            __builtin_prefetch(krow + (size_t)ROWS_PER_TILE * D, 0, 0);
            __builtin_prefetch(krow + (size_t)ROWS_PER_TILE * D + 64, 0, 0);
        }

        // ----- load k directly in A-fragment order, compute row norms -----
        // lane covers K = kb*32 + 8*hi + {0..7} and + {16..23}; the (hi=0,hi=1)
        // pair exactly tiles the 128-wide row.
        float av[4][16];
        float ss = 0.0f;
#pragma unroll
        for (int kb = 0; kb < 4; ++kb) {
            const float4* p0 = (const float4*)(krow + kb * 32 + hi * 8);
            const float4* p1 = (const float4*)(krow + kb * 32 + hi * 8 + 16);
            float4 x0 = p0[0], x1 = p0[1], y0 = p1[0], y1 = p1[1];
            av[kb][0]  = x0.x; av[kb][1]  = x0.y; av[kb][2]  = x0.z; av[kb][3]  = x0.w;
            av[kb][4]  = x1.x; av[kb][5]  = x1.y; av[kb][6]  = x1.z; av[kb][7]  = x1.w;
            av[kb][8]  = y0.x; av[kb][9]  = y0.y; av[kb][10] = y0.z; av[kb][11] = y0.w;
            av[kb][12] = y1.x; av[kb][13] = y1.y; av[kb][14] = y1.z; av[kb][15] = y1.w;
#pragma unroll
            for (int e = 0; e < 16; ++e) ss = fmaf(av[kb][e], av[kb][e], ss);
        }
        ss += __shfl_xor(ss, 16);                // combine the row's lane pair
        const float nrm  = sqrtf(ss);
        const float rinv = 1.0f / (nrm + 1e-10f);
        if (hi == 0) Nrm[wid][m16] = nrm;

        Frag16 a1[4];
#pragma unroll
        for (int kb = 0; kb < 4; ++kb)
#pragma unroll
            for (int e = 0; e < 16; ++e)
                a1[kb].h[e] = (_Float16)(av[kb][e] * rinv);

        // ---- GEMM1: shaped = k_unit @ (cs*R)^T, then snap to centroid ----
#pragma unroll
        for (int nb = 0; nb < 8; ++nb) {
            const int n = nb * 16 + m16;         // this lane's output column
            v8f c = {};
#pragma unroll
            for (int kb = 0; kb < 4; ++kb) {
                // B1[K][N] = cs[n]*R[n][K] -> Rh row n, K = kb*32 + 8*hi + {0..7, 16..23}
                Frag16 b;
                const uint4* bp = (const uint4*)(Rh + n * D + kb * 32 + hi * 8);
                b.u[0] = bp[0]; b.u[1] = bp[2];
                c = __builtin_amdgcn_wmma_f32_16x16x32_f16(
                        false, a1[kb].v, false, b.v, (short)0, c, false, false);
            }
            // quantize: element v sits at row m = 8*hi + v, column n
#pragma unroll
            for (int v = 0; v < 8; ++v) {
                const float s = c[v];            // cs already folded into Rh
                float z = cents[0];
#pragma unroll
                for (int j = 0; j < NCENT - 1; ++j)
                    z = (s > bnd[j]) ? cents[j + 1] : z;   // searchsorted(left)
                Qs[wid][(8 * hi + v) * D + n] = (_Float16)z;
            }
        }

        // ---------- GEMM2: k_hat = z @ (inv*R), scale by norms ----------
        Frag16 a2[4];
#pragma unroll
        for (int kb = 0; kb < 4; ++kb) {         // same A layout, from staging
            const _Float16* s = &Qs[wid][m16 * D + kb * 32 + hi * 8];
            a2[kb].u[0] = *(const uint4*)(s);
            a2[kb].u[1] = *(const uint4*)(s + 16);
        }
        float nm[8];
#pragma unroll
        for (int v = 0; v < 8; ++v) nm[v] = Nrm[wid][8 * hi + v];

#pragma unroll
        for (int nb = 0; nb < 8; ++nb) {
            const int n = nb * 16 + m16;
            v8f c = {};
#pragma unroll
            for (int kb = 0; kb < 4; ++kb) {
                // B2[K][N] = inv[K]*R[K][n] = RhT[n][K] -> K = kb*32 + 8*hi + {0..7, 16..23}
                Frag16 b;
                const uint4* bp = (const uint4*)(RhT + n * D + kb * 32 + hi * 8);
                b.u[0] = bp[0]; b.u[1] = bp[2];
                c = __builtin_amdgcn_wmma_f32_16x16x32_f16(
                        false, a2[kb].v, false, b.v, (short)0, c, false, false);
            }
#pragma unroll
            for (int v = 0; v < 8; ++v)
                out[(row0 + 8 * hi + v) * D + n] = c[v] * nm[v];
        }
    }
}

extern "C" void kernel_launch(void* const* d_in, const int* in_sizes, int n_in,
                              void* d_out, int out_size, void* d_ws, size_t ws_size,
                              hipStream_t stream) {
    const float* k      = (const float*)d_in[0];
    const float* rot    = (const float*)d_in[1];
    const float* cscale = (const float*)d_in[2];
    const float* cent   = (const float*)d_in[3];
    float* out          = (float*)d_out;

    const int nrows = in_sizes[0] / D;                          // 1048576
    dim3 grid(nrows / (ROWS_PER_TILE * TILES_PER_BLOCK));       // 1024 blocks
    dim3 block(256);                                            // 8 wave32
    hipLaunchKernelGGL(rotquant_fused_kernel, grid, block, 0, stream,
                       k, rot, cscale, cent, out);
}
